// VQVAEModel_24902220382360
// MI455X (gfx1250) — matrix operand order repaired and checked by reading, not compiled, not run on hardware
//
#include <hip/hip_runtime.h>
#include <math.h>

// ---------------------------------------------------------------------------
// VQ-VAE forward for MI455X (gfx1250, wave32, WMMA).
// Heavy GEMMs (enc conv2, VQ distance, dec convT1) use v_wmma_f32_16x16x32_bf16
// with f32 accumulation; cheap low-intensity convs (conv1, convT2) use VALU.
// ---------------------------------------------------------------------------

typedef __bf16  bf16;
typedef __attribute__((ext_vector_type(16))) __bf16 v16bf;
typedef __attribute__((ext_vector_type(8)))  float  v8f;

union V16 { v16bf v; unsigned u[8]; };

__device__ __forceinline__ int kmap_a(int h, int lane) {
  // 16-bit A 16x32 layout: lanes0-15 rows, K in {0..7,16..23}; lanes16-31 K+8
  return ((h & 8) << 1) + (h & 7) + ((lane & 16) ? 8 : 0);
}
__device__ __forceinline__ int kmap_b(int h, int lane) {
  // 16-bit B 32x16 layout: lanes0-15 K in {0..7,16..23}; lanes16-31 K+8
  return ((h & 8) << 1) + (h & 7) + ((lane & 16) ? 8 : 0);
}
__device__ __forceinline__ v8f wmma_bf16(v16bf a, v16bf b, v8f c) {
  return __builtin_amdgcn_wmma_f32_16x16x32_bf16(false, a, false, b,
                                                 (short)0, c, false, false);
}

// ----------------------------- constants -----------------------------------
#define BATCH 128
#define HID   128
#define DIM   256
#define KCB   1024
#define NPIX  32768           // BATCH*16*16 latent pixels
#define XHAT_ELEMS 1572864    // 128*3*64*64
#define LOSS_OFF   1572864
#define PERP_OFF   1572865
#define OH_OFF     1572866

// ----------------------------- init ----------------------------------------
__global__ void k_init(float* counts, float* lossacc) {
  int t = threadIdx.x;
  if (t < KCB) counts[t] = 0.f;
  if (t == 0)  *lossacc = 0.f;
}

// ------------------------- weight prep (bf16, A-ready) ----------------------
// enc_w2 (O=256,I=128,4,4) -> W2a[o][tap*128+c]
__global__ void k_prep_w2(const float* __restrict__ w, bf16* __restrict__ out) {
  int t = blockIdx.x * 256 + threadIdx.x;            // 524288
  int o = t >> 11, rem = t & 2047, tap = rem >> 7, c = rem & 127;
  out[t] = (bf16)w[o * 2048 + c * 16 + tap];
}

// dec_w1 (I=256,O=128,4,4) -> Wc[class][o][j*256+c], class=(py,px), 4 taps each
__global__ void k_prep_w1d(const float* __restrict__ w, bf16* __restrict__ out) {
  int t = blockIdx.x * 256 + threadIdx.x;            // 524288
  int cls = t >> 17, rem = t & 131071;
  int o = rem >> 10, r2 = rem & 1023, j = r2 >> 8, c = r2 & 255;
  int py = cls >> 1, px = cls & 1, jy = j >> 1, jx = j & 1;
  int kh = py ? 2 * jy : 1 + 2 * jy;
  int kw = px ? 2 * jx : 1 + 2 * jx;
  out[t] = (bf16)w[c * 2048 + o * 16 + kh * 4 + kw];
}

// codebook f32 -> bf16 copy + per-code squared norms
__global__ void k_prep_cb(const float* __restrict__ cb, bf16* __restrict__ cbh,
                          float* __restrict__ cnorm) {
  __shared__ float red[256];
  int j = blockIdx.x, c = threadIdx.x;
  float v = cb[j * 256 + c];
  cbh[j * 256 + c] = (bf16)v;
  red[c] = v * v;
  __syncthreads();
  for (int s = 128; s > 0; s >>= 1) { if (c < s) red[c] += red[c + s]; __syncthreads(); }
  if (c == 0) cnorm[j] = red[0];
}

// ------------------------- conv1 (VALU, K=48) -------------------------------
// x (B,3,64,64) -> h BHWC bf16 (B,32,32,128), ReLU
__global__ __launch_bounds__(128) void k_conv1(const float* __restrict__ x,
                                               const float* __restrict__ w,
                                               const float* __restrict__ bias,
                                               bf16* __restrict__ h) {
  __shared__ float patch[48];
  int blk = blockIdx.x;                // pixel
  int bi = blk >> 10, rem = blk & 1023, y = rem >> 5, xx = rem & 31;
  int t = threadIdx.x;                 // out channel
  if (t < 48) {
    int c = t >> 4, kh = (t >> 2) & 3, kw = t & 3;
    int iy = 2 * y - 1 + kh, ix = 2 * xx - 1 + kw;
    patch[t] = ((unsigned)iy < 64u && (unsigned)ix < 64u)
                   ? x[((bi * 3 + c) * 64 + iy) * 64 + ix] : 0.f;
  }
  __syncthreads();
  float acc = bias[t];
  const float* wr = w + t * 48;
#pragma unroll
  for (int i = 0; i < 48; i++) acc += patch[i] * wr[i];
  h[blk * 128 + t] = (bf16)fmaxf(acc, 0.f);
}

// ------------------------- conv2 (WMMA) -------------------------------------
// h BHWC bf16 -> e BHWC f32 (B,16,16,256). M=256, N=32768, K=2048.
__global__ __launch_bounds__(32) void k_conv2(const bf16* __restrict__ h,
                                              const bf16* __restrict__ W2a,
                                              const float* __restrict__ b2,
                                              float* __restrict__ e) {
  const int lane = threadIdx.x;
  const int g = blockIdx.x;        // 0..255  (8 N-tiles each)
  const int mt = blockIdx.y;       // 0..15
  const int o = mt * 16 + (lane & 15);
  const bf16* rowA = W2a + o * 2048;

  v8f acc[8];
  v8f z8 = {0.f, 0.f, 0.f, 0.f, 0.f, 0.f, 0.f, 0.f};
#pragma unroll
  for (int nb = 0; nb < 8; nb++) acc[nb] = z8;

  int pb[8], py_[8], px_[8];
#pragma unroll
  for (int nb = 0; nb < 8; nb++) {
    int p = g * 128 + nb * 16 + (lane & 15);
    pb[nb] = p >> 8; py_[nb] = (p >> 4) & 15; px_[nb] = p & 15;
  }

  for (int tap = 0; tap < 16; ++tap) {
    const int kh = tap >> 2, kw = tap & 3;
    const bf16* bptr[8]; bool valid[8];
#pragma unroll
    for (int nb = 0; nb < 8; nb++) {
      int iy = 2 * py_[nb] - 1 + kh, ix = 2 * px_[nb] - 1 + kw;
      valid[nb] = ((unsigned)iy < 32u) && ((unsigned)ix < 32u);
      bptr[nb] = h + ((pb[nb] * 32 + iy) * 32 + ix) * 128;
    }
#pragma unroll
    for (int cc = 0; cc < 4; ++cc) {
      V16 A;
#pragma unroll
      for (int i = 0; i < 8; i++) {
        int k = tap * 128 + cc * 32 + kmap_a(2 * i, lane);
        A.u[i] = *reinterpret_cast<const unsigned*>(rowA + k);
      }
#pragma unroll
      for (int nb = 0; nb < 8; nb++) {
        V16 B;
        if (valid[nb]) {
#pragma unroll
          for (int i = 0; i < 8; i++) {
            int c = cc * 32 + kmap_b(2 * i, lane);
            B.u[i] = *reinterpret_cast<const unsigned*>(bptr[nb] + c);
          }
        } else {
#pragma unroll
          for (int i = 0; i < 8; i++) B.u[i] = 0u;
        }
        acc[nb] = wmma_bf16(A.v, B.v, acc[nb]);
      }
    }
  }
  float bias[8];
#pragma unroll
  for (int r = 0; r < 8; r++) {
    int m = r + ((lane & 16) ? 8 : 0);
    bias[r] = b2[mt * 16 + m];
  }
#pragma unroll
  for (int nb = 0; nb < 8; nb++) {
    int p = g * 128 + nb * 16 + (lane & 15);
#pragma unroll
    for (int r = 0; r < 8; r++) {
      int m = r + ((lane & 16) ? 8 : 0);
      e[p * 256 + mt * 16 + m] = acc[nb][r] + bias[r];
    }
  }
}

// ------------------------- VQ argmin (WMMA) ---------------------------------
// e BHWC f32 rows vs codebook bf16: fused dist + argmin. One wave = 16 pixels.
__global__ __launch_bounds__(32) void k_vq(const float* __restrict__ e,
                                           const bf16* __restrict__ cb,
                                           const float* __restrict__ cnorm,
                                           int* __restrict__ idx) {
  __shared__ float part[32];
  __shared__ float dtile[256];
  const int lane = threadIdx.x;
  const int p0 = blockIdx.x * 16;
  const int pix = p0 + (lane & 15);
  const float* rowE = e + pix * 256;

  V16 A[8];
  float xpart = 0.f;
#pragma unroll
  for (int kc = 0; kc < 8; kc++) {
#pragma unroll
    for (int i = 0; i < 8; i++) {
      int k = kc * 32 + kmap_a(2 * i, lane);
      float2 f = *reinterpret_cast<const float2*>(rowE + k);
      A[kc].v[2 * i]     = (bf16)f.x;
      A[kc].v[2 * i + 1] = (bf16)f.y;
      xpart += f.x * f.x + f.y * f.y;
    }
  }
  part[lane] = xpart;
  __syncthreads();
  float xnf[8];
#pragma unroll
  for (int r = 0; r < 8; r++) {
    int m = r + ((lane & 16) ? 8 : 0);
    xnf[r] = part[m] + part[m + 16];
  }

  float best = 3.4e38f; int bestI = 0;
  v8f z8 = {0.f, 0.f, 0.f, 0.f, 0.f, 0.f, 0.f, 0.f};
  for (int t = 0; t < 64; t++) {
    v8f acc = z8;
    const bf16* rowB = cb + (t * 16 + (lane & 15)) * 256;
#pragma unroll
    for (int kc = 0; kc < 8; kc++) {
      V16 B;
#pragma unroll
      for (int i = 0; i < 8; i++) {
        int k = kc * 32 + kmap_b(2 * i, lane);
        B.u[i] = *reinterpret_cast<const unsigned*>(rowB + k);
      }
      acc = wmma_bf16(A[kc].v, B.v, acc);
    }
    float cn = cnorm[t * 16 + (lane & 15)];
    __syncthreads();
#pragma unroll
    for (int r = 0; r < 8; r++) {
      int m = r + ((lane & 16) ? 8 : 0);
      dtile[m * 16 + (lane & 15)] = xnf[r] + cn - 2.0f * acc[r];
    }
    __syncthreads();
    if (lane < 16) {
#pragma unroll
      for (int n = 0; n < 16; n++) {
        float d = dtile[lane * 16 + n];
        if (d < best) { best = d; bestI = t * 16 + n; }   // first-min tie rule
      }
    }
  }
  if (lane < 16) idx[pix] = bestI;
}

// ------------------ quantize + commitment loss + histogram ------------------
__global__ void k_quant(const int* __restrict__ idx, const float* __restrict__ cb,
                        const float* __restrict__ e, bf16* __restrict__ q,
                        float* __restrict__ lossacc, float* __restrict__ counts) {
  __shared__ float red[256];
  int row = blockIdx.x, c = threadIdx.x;
  int j = idx[row];
  float qv = cb[j * 256 + c];
  float ev = e[row * 256 + c];
  q[row * 256 + c] = (bf16)qv;
  float d = qv - ev;
  red[c] = d * d;
  __syncthreads();
  for (int s = 128; s > 0; s >>= 1) { if (c < s) red[c] += red[c + s]; __syncthreads(); }
  if (c == 0) { atomicAdd(lossacc, red[0]); atomicAdd(&counts[j], 1.0f); }
}

// ------------------------- one-hot encodings (B,K,256) ----------------------
__global__ void k_onehot(const int* __restrict__ idx, float* __restrict__ out) {
  int gid = blockIdx.x * 256 + threadIdx.x;   // 33,554,432 total
  int b = gid >> 18, rem = gid & 262143;
  int k = rem >> 8, n = rem & 255;
  out[gid] = (idx[b * 256 + n] == k) ? 1.0f : 0.0f;
}

// ------------------------- dec convT1 (WMMA) --------------------------------
// q BHWC bf16 (B,16,16,256) -> d1 BHWC bf16 (B,32,32,128), ReLU. 4 parity classes.
__global__ __launch_bounds__(32) void k_dect1(const bf16* __restrict__ q,
                                              const bf16* __restrict__ Wc,
                                              const float* __restrict__ b1,
                                              bf16* __restrict__ d1) {
  const int lane = threadIdx.x;
  const int g = blockIdx.x;        // 0..255
  const int mt = blockIdx.y;       // 0..7
  const int cls = blockIdx.z;      // 0..3
  const int py = cls >> 1, px = cls & 1;
  const int o = mt * 16 + (lane & 15);
  const bf16* rowA = Wc + cls * 131072 + o * 1024;

  v8f acc[8];
  v8f z8 = {0.f, 0.f, 0.f, 0.f, 0.f, 0.f, 0.f, 0.f};
#pragma unroll
  for (int nb = 0; nb < 8; nb++) acc[nb] = z8;

  int pb[8], pty[8], ptx[8];
#pragma unroll
  for (int nb = 0; nb < 8; nb++) {
    int n = g * 128 + nb * 16 + (lane & 15);
    pb[nb] = n >> 8; pty[nb] = (n >> 4) & 15; ptx[nb] = n & 15;
  }

  for (int j = 0; j < 4; j++) {
    int jy = j >> 1, jx = j & 1;
    int dy = (py == 0) ? -jy : 1 - jy;
    int dx = (px == 0) ? -jx : 1 - jx;
    const bf16* bptr[8]; bool valid[8];
#pragma unroll
    for (int nb = 0; nb < 8; nb++) {
      int iy = pty[nb] + dy, ix = ptx[nb] + dx;
      valid[nb] = ((unsigned)iy < 16u) && ((unsigned)ix < 16u);
      bptr[nb] = q + ((pb[nb] * 16 + iy) * 16 + ix) * 256;
    }
#pragma unroll
    for (int cc = 0; cc < 8; ++cc) {
      V16 A;
#pragma unroll
      for (int i = 0; i < 8; i++) {
        int k = j * 256 + cc * 32 + kmap_a(2 * i, lane);
        A.u[i] = *reinterpret_cast<const unsigned*>(rowA + k);
      }
#pragma unroll
      for (int nb = 0; nb < 8; nb++) {
        V16 B;
        if (valid[nb]) {
#pragma unroll
          for (int i = 0; i < 8; i++) {
            int c = cc * 32 + kmap_b(2 * i, lane);
            B.u[i] = *reinterpret_cast<const unsigned*>(bptr[nb] + c);
          }
        } else {
#pragma unroll
          for (int i = 0; i < 8; i++) B.u[i] = 0u;
        }
        acc[nb] = wmma_bf16(A.v, B.v, acc[nb]);
      }
    }
  }
  float bias[8];
#pragma unroll
  for (int r = 0; r < 8; r++) {
    int m = r + ((lane & 16) ? 8 : 0);
    bias[r] = b1[mt * 16 + m];
  }
#pragma unroll
  for (int nb = 0; nb < 8; nb++) {
    int oy = 2 * pty[nb] + py, ox = 2 * ptx[nb] + px;
    long pout = (long)((pb[nb] * 32 + oy) * 32 + ox) * 128;
#pragma unroll
    for (int r = 0; r < 8; r++) {
      int m = r + ((lane & 16) ? 8 : 0);
      d1[pout + mt * 16 + m] = (bf16)fmaxf(acc[nb][r] + bias[r], 0.f);
    }
  }
}

// ------------------------- dec convT2 (VALU) --------------------------------
// d1 BHWC bf16 -> x_hat (B,3,64,64) f32
__global__ __launch_bounds__(256) void k_dect2(const bf16* __restrict__ d1,
                                               const float* __restrict__ w,
                                               const float* __restrict__ b2,
                                               float* __restrict__ out) {
  __shared__ float wl[2048];           // [c][kh*4+kw] for this co
  int plane = blockIdx.x >> 4;         // b*3+co
  int co = plane % 3, b = plane / 3;
  for (int t = threadIdx.x; t < 2048; t += 256) {
    int c = t >> 4, kk = t & 15;
    wl[t] = w[c * 48 + co * 16 + kk];
  }
  __syncthreads();
  int r2 = (blockIdx.x & 15) * 256 + threadIdx.x;   // 0..4095
  int oy = r2 >> 6, ox = r2 & 63;
  float acc = b2[co];
  int kh0 = (oy + 1) & 1, kw0 = (ox + 1) & 1;
  for (int dh = 0; dh < 2; ++dh) {
    int kh = kh0 + 2 * dh, iy = (oy + 1 - kh) >> 1;
    if ((unsigned)iy >= 32u) continue;
    for (int dw = 0; dw < 2; ++dw) {
      int kw = kw0 + 2 * dw, ix = (ox + 1 - kw) >> 1;
      if ((unsigned)ix >= 32u) continue;
      const bf16* src = d1 + ((b * 32 + iy) * 32 + ix) * 128;
      const float* wp = wl + kh * 4 + kw;
#pragma unroll 8
      for (int c = 0; c < 128; c++) acc += (float)src[c] * wp[c * 16];
    }
  }
  out[(long)plane * 4096 + r2] = acc;
}

// ------------------------- finalize loss + perplexity -----------------------
__global__ void k_final(const float* __restrict__ counts,
                        const float* __restrict__ lossacc,
                        float* __restrict__ out) {
  __shared__ float red[1024];
  int k = threadIdx.x;
  float p = counts[k] * (1.0f / 32768.0f);
  red[k] = p * logf(p + 1e-10f);
  __syncthreads();
  for (int s = 512; s > 0; s >>= 1) { if (k < s) red[k] += red[k + s]; __syncthreads(); }
  if (k == 0) {
    out[PERP_OFF] = expf(-red[0]);
    out[LOSS_OFF] = 0.25f * (*lossacc) * (1.0f / 8388608.0f);
  }
}

// ---------------------------------------------------------------------------
extern "C" void kernel_launch(void* const* d_in, const int* in_sizes, int n_in,
                              void* d_out, int out_size, void* d_ws, size_t ws_size,
                              hipStream_t stream) {
  const float* x      = (const float*)d_in[0];
  const float* enc_w1 = (const float*)d_in[1];
  const float* enc_b1 = (const float*)d_in[2];
  const float* enc_w2 = (const float*)d_in[3];
  const float* enc_b2 = (const float*)d_in[4];
  const float* dec_w1 = (const float*)d_in[5];
  const float* dec_b1 = (const float*)d_in[6];
  const float* dec_w2 = (const float*)d_in[7];
  const float* dec_b2 = (const float*)d_in[8];
  const float* cbook  = (const float*)d_in[9];
  float* out = (float*)d_out;

  char* ws = (char*)d_ws;
  size_t off = 0;
  bf16*  h_bhwc = (bf16*)(ws + off);  off += 33554432;  // B*32*32*128 bf16
  float* e_bhwc = (float*)(ws + off); off += 33554432;  // B*16*16*256 f32
  bf16*  q_bhwc = (bf16*)(ws + off);  off += 16777216;  // B*16*16*256 bf16
  bf16*  d1_bhwc= (bf16*)(ws + off);  off += 33554432;  // B*32*32*128 bf16
  bf16*  W2a    = (bf16*)(ws + off);  off += 1048576;   // 256x2048 bf16
  bf16*  W1d    = (bf16*)(ws + off);  off += 1048576;   // 4x128x1024 bf16
  bf16*  cb_bf  = (bf16*)(ws + off);  off += 524288;    // 1024x256 bf16
  float* cnorm  = (float*)(ws + off); off += 4096;
  float* counts = (float*)(ws + off); off += 4096;
  float* lossac = (float*)(ws + off); off += 16;
  int*   idx    = (int*)(ws + off);   off += 131072;

  k_init    <<<1, 1024, 0, stream>>>(counts, lossac);
  k_prep_w2 <<<2048, 256, 0, stream>>>(enc_w2, W2a);
  k_prep_w1d<<<2048, 256, 0, stream>>>(dec_w1, W1d);
  k_prep_cb <<<1024, 256, 0, stream>>>(cbook, cb_bf, cnorm);

  k_conv1   <<<131072, 128, 0, stream>>>(x, enc_w1, enc_b1, h_bhwc);
  k_conv2   <<<dim3(256, 16), 32, 0, stream>>>(h_bhwc, W2a, enc_b2, e_bhwc);
  k_vq      <<<2048, 32, 0, stream>>>(e_bhwc, cb_bf, cnorm, idx);
  k_quant   <<<32768, 256, 0, stream>>>(idx, cbook, e_bhwc, q_bhwc, lossac, counts);
  k_onehot  <<<131072, 256, 0, stream>>>(idx, out + OH_OFF);
  k_dect1   <<<dim3(256, 8, 4), 32, 0, stream>>>(q_bhwc, W1d, dec_b1, d1_bhwc);
  k_dect2   <<<6144, 256, 0, stream>>>(d1_bhwc, dec_w2, dec_b2, out);
  k_final   <<<1, 1024, 0, stream>>>(counts, lossac, out);
}
